// KGAT_85323820302857
// MI455X (gfx1250) — compile-verified, matrix operand order checked by reading, not run on hardware
//
#include <hip/hip_runtime.h>
#include <hip/hip_bf16.h>

// ---------------------------------------------------------------------------
// KGAT forward on gfx1250 (CDNA5, wave32, WMMA bf16 16x16x32 w/ f32 accum,
// LDS staging with async global->LDS copies where available)
// ---------------------------------------------------------------------------

typedef __attribute__((ext_vector_type(16))) __bf16 v16bf;
typedef __attribute__((ext_vector_type(8)))  __bf16 v8bf;
typedef __attribute__((ext_vector_type(8)))  float  v8f;
typedef __attribute__((ext_vector_type(4)))  float  v4f;

#define KG_N_USERS   30000
#define KG_N_NODES   100000
#define KG_N_REL     40
#define KG_D         64
#define KG_OUT_DIM   160          // 64 + 64 + 32 concatenated
#define KG_NEG_SLOPE 0.01f

// gfx1250 async global->LDS copy path (guarded: zero compile risk elsewhere)
#if defined(__AMDGCN__) && __has_builtin(__builtin_amdgcn_global_load_async_to_lds_b128)
#define KG_ASYNC 1
#else
#define KG_ASYNC 0
#endif

#define KG_ASG __attribute__((address_space(1)))
#define KG_ASL __attribute__((address_space(3)))

#if KG_ASYNC
// builtin expects int4 (vector_size(16)) pointers: AS1 source, AS3 LDS dest
typedef int kg_v4i __attribute__((vector_size(16)));
#define KG_GP(p) ((KG_ASG kg_v4i*)(p))
#define KG_LP(p) ((KG_ASL kg_v4i*)(p))
#endif

__device__ __forceinline__ void kg_wait_async0() {
#if defined(__AMDGCN__)
#if __has_builtin(__builtin_amdgcn_s_wait_asynccnt)
  __builtin_amdgcn_s_wait_asynccnt(0);
#else
  asm volatile("s_wait_asynccnt 0x0" ::: "memory");
#endif
#endif
}

// float -> bf16, round-to-nearest-even
__device__ __forceinline__ __bf16 f2bf(float f) {
  unsigned u = __float_as_uint(f);
  unsigned r = (u + 0x7FFFu + ((u >> 16) & 1u)) >> 16;
  unsigned short s = (unsigned short)r;
  __bf16 b;
  __builtin_memcpy(&b, &s, 2);
  return b;
}

// order-preserving float->uint key for atomic max over signed floats
__device__ __forceinline__ unsigned f2key(float f) {
  unsigned u = __float_as_uint(f);
  return (u & 0x80000000u) ? ~u : (u | 0x80000000u);
}
__device__ __forceinline__ float key2f(unsigned k) {
  unsigned u = (k & 0x80000000u) ? (k & 0x7FFFFFFFu) : ~k;
  return __uint_as_float(u);
}

// ---------------------------------------------------------------------------
// Prep kernels
// ---------------------------------------------------------------------------

__global__ void k_zero_f32(float* __restrict__ p, long long n) {
  long long i = (long long)blockIdx.x * blockDim.x + threadIdx.x;
  if (i < n) p[i] = 0.0f;
}

__global__ void k_f32_to_bf16(const float* __restrict__ s, __bf16* __restrict__ d, long long n) {
  long long i = (long long)blockIdx.x * blockDim.x + threadIdx.x;
  if (i < n) d[i] = f2bf(s[i]);
}

// Pre-swizzle [K=64 x Nout] weight matrices (row-major, Nout contiguous) into
// the wave32 WMMA B-operand lane layout: per (colTile ct, kTile kt), lane l
// holds 16 contiguous bf16 for column N = ct*16 + (l&15), K per ISA layout.
__global__ void k_pack_b(const float* __restrict__ W, __bf16* __restrict__ WB,
                         int Nout, int nMat) {
  long long total = (long long)nMat * 64 * Nout;
  long long tid = (long long)blockIdx.x * blockDim.x + threadIdx.x;
  if (tid >= total) return;
  int perMat = 64 * Nout;
  int mat = (int)(tid / perMat);
  int rem = (int)(tid % perMat);
  int ct   = rem >> 10;            // / 1024  (2 kTiles * 32 lanes * 16)
  int r2   = rem & 1023;
  int kt   = r2 >> 9;              // / 512
  int r3   = r2 & 511;
  int lane = r3 >> 4;
  int j    = r3 & 15;
  int half = lane >> 4;
  int nc   = ct * 16 + (lane & 15);
  int K    = kt * 32 + ((j < 8) ? (half * 8 + j) : (16 + half * 8 + (j - 8)));
  WB[(long long)mat * perMat + rem] = f2bf(W[(long long)mat * perMat + K * Nout + nc]);
}

// ---------------------------------------------------------------------------
// Relation projection: P[r] = emb @ W_R[r]  (bf16 WMMA, f32 accum)
// block = 256 (8 waves); wave w: colTile = w&3, rowSubTile = w>>2.
// A (32 rows) and B (relation, 8KB) staged in LDS (async path if available);
// output bounced through per-wave LDS patch for coalesced NT b128 stores.
// grid = (3125, KG_N_REL)
// ---------------------------------------------------------------------------
__global__ __launch_bounds__(256)
void k_proj(const __bf16* __restrict__ embB, const __bf16* __restrict__ WBrel,
            float* __restrict__ P) {
  __shared__ __bf16 sB[4096];        // 8 KB: packed B for this relation
  __shared__ __bf16 sA[2048];        // 4 KB: 32 emb rows
  __shared__ float  sOut[8][256];    // 8 KB: per-wave 16x16 output patch

  int tid  = threadIdx.x;
  int lane = tid & 31;
  int wave = tid >> 5;
  int r    = blockIdx.y;
  long long rowBase = (long long)blockIdx.x * 32;

  const __bf16* gB = WBrel + (long long)r * 4096;
  const __bf16* gA = embB + rowBase * 64;

#if KG_ASYNC
  __builtin_amdgcn_global_load_async_to_lds_b128(
      KG_GP(gB + tid * 16),     KG_LP(sB + tid * 16), 0, 0);
  __builtin_amdgcn_global_load_async_to_lds_b128(
      KG_GP(gB + tid * 16 + 8), KG_LP(sB + tid * 16 + 8), 0, 0);
  __builtin_amdgcn_global_load_async_to_lds_b128(
      KG_GP(gA + tid * 8),      KG_LP(sA + tid * 8), 0, 0);
  kg_wait_async0();
#else
  {
    uint4*       db = (uint4*)sB;                 // 512 uint4
    const uint4* sb = (const uint4*)gB;
    db[tid * 2]     = sb[tid * 2];
    db[tid * 2 + 1] = sb[tid * 2 + 1];
    ((uint4*)sA)[tid] = ((const uint4*)gA)[tid];  // 256 uint4
  }
#endif
  __syncthreads();

  int ct   = wave & 3;
  int rsub = wave >> 2;
  int half = lane >> 4;
  int mlo  = lane & 15;
  int rowL = rsub * 16 + mlo;

  v8f acc = {};
#pragma unroll
  for (int kt = 0; kt < 2; ++kt) {
    int kk = kt * 32;
    v8bf c0 = *(const v8bf*)(sA + rowL * 64 + kk + half * 8);
    v8bf c1 = *(const v8bf*)(sA + rowL * 64 + kk + 16 + half * 8);
    v16bf a;
#pragma unroll
    for (int i = 0; i < 8; ++i) { a[i] = c0[i]; a[i + 8] = c1[i]; }
    v16bf b = *(const v16bf*)(sB + ct * 1024 + kt * 512 + lane * 16);
    acc = __builtin_amdgcn_wmma_f32_16x16x32_bf16(false, a, false, b,
                                                  (short)0, acc, false, false);
  }

  // transpose D layout through per-wave LDS patch -> coalesced NT stores
  float* so = sOut[wave];
#pragma unroll
  for (int i = 0; i < 8; ++i) so[(half * 8 + i) * 16 + mlo] = acc[i];
  int rrow = lane >> 1;
  int c8   = (lane & 1) * 8;
  v4f o0 = *(const v4f*)(so + rrow * 16 + c8);
  v4f o1 = *(const v4f*)(so + rrow * 16 + c8 + 4);
  float* gp = P + (((long long)r * KG_N_NODES + rowBase + rsub * 16 + rrow) * 64
                   + ct * 16 + c8);
  __builtin_nontemporal_store(o0, (v4f*)gp);
  __builtin_nontemporal_store(o1, (v4f*)(gp + 4));
}

// ---------------------------------------------------------------------------
// Edge attention: att[e] = sum_k P[r][src][k] * tanh(P[r][dst][k] + rel[r][k])
// one wave32 per edge; running segment max via ordered-uint atomicMax.
// ---------------------------------------------------------------------------
__global__ __launch_bounds__(256)
void k_att(const float* __restrict__ P, const float* __restrict__ relE,
           const int* __restrict__ src, const int* __restrict__ dst,
           const int* __restrict__ et, float* __restrict__ att,
           unsigned* __restrict__ segmaxKey, int E) {
  int e    = blockIdx.x * 8 + (threadIdx.x >> 5);
  int lane = threadIdx.x & 31;
  if (e >= E) return;
  int s = src[e], d = dst[e], r = et[e];
  const float* Ps = P + ((long long)r * KG_N_NODES + s) * 64;
  const float* Pd = P + ((long long)r * KG_N_NODES + d) * 64;
  const float* re = relE + r * 64;
  float sum = 0.f;
#pragma unroll
  for (int j = 0; j < 2; ++j) {
    int k = lane + 32 * j;
    sum += Ps[k] * tanhf(Pd[k] + re[k]);
  }
#pragma unroll
  for (int m = 16; m; m >>= 1) sum += __shfl_xor(sum, m, 32);
  if (lane == 0) {
    att[e] = sum;
    atomicMax(segmaxKey + d, f2key(sum));
  }
}

__global__ void k_expsum(float* __restrict__ att, const unsigned* __restrict__ segmaxKey,
                         const int* __restrict__ dst, float* __restrict__ segsum, int E) {
  int e = blockIdx.x * blockDim.x + threadIdx.x;
  if (e >= E) return;
  int d = dst[e];
  float ex = __expf(att[e] - key2f(segmaxKey[d]));
  att[e] = ex;                      // att now holds exp(att - max)
  atomicAdd(segsum + d, ex);
}

// ---------------------------------------------------------------------------
// Message passing: Nh[dst] += x[src] * (ex / segsum[dst]); one wave per edge.
// ---------------------------------------------------------------------------
__global__ __launch_bounds__(256)
void k_msg(const float* __restrict__ x, const float* __restrict__ att,
           const float* __restrict__ segsum, const int* __restrict__ src,
           const int* __restrict__ dst, float* __restrict__ Nh, int E) {
  int e    = blockIdx.x * 8 + (threadIdx.x >> 5);
  int lane = threadIdx.x & 31;
  if (e >= E) return;
  int s = src[e], d = dst[e];
  float w = att[e] / segsum[d];
#pragma unroll
  for (int j = 0; j < 2; ++j) {
    int k = lane + 32 * j;
    atomicAdd(Nh + (long long)d * 64 + k, x[(long long)s * 64 + k] * w);
  }
}

// ---------------------------------------------------------------------------
// Bi-interaction layer: out = leaky((x+Nh)@W1+b1) + leaky((x*Nh)@W2+b2)
// A operands built on the fly in bf16 regs (vectorized f32 loads); outputs
// bounced through a per-wave LDS patch for coalesced b128 stores.
// ---------------------------------------------------------------------------
template <int DO>
__global__ __launch_bounds__(256)
void k_bi(const float* __restrict__ x, const float* __restrict__ Nh,
          const __bf16* __restrict__ W1B, const float* __restrict__ b1,
          const __bf16* __restrict__ W2B, const float* __restrict__ b2,
          float* __restrict__ out) {
  __shared__ float sOut[8][256];
  constexpr int CT = DO / 16;       // col tiles
  constexpr int RT = 8 / CT;        // row sub-tiles per block
  int lane = threadIdx.x & 31;
  int wave = threadIdx.x >> 5;
  int ct   = wave % CT;
  int nb   = (blockIdx.x * RT + wave / CT) * 16;
  if (nb >= KG_N_NODES) return;
  int half = lane >> 4;
  int mlo  = lane & 15;
  long long row = nb + mlo;

  v8f accS = {}, accP = {};
#pragma unroll
  for (int kt = 0; kt < 2; ++kt) {
    int k0 = kt * 32 + half * 8;
    int k1 = kt * 32 + 16 + half * 8;
    v4f x0a = *(const v4f*)(x  + row * 64 + k0);
    v4f x0b = *(const v4f*)(x  + row * 64 + k0 + 4);
    v4f n0a = *(const v4f*)(Nh + row * 64 + k0);
    v4f n0b = *(const v4f*)(Nh + row * 64 + k0 + 4);
    v4f x1a = *(const v4f*)(x  + row * 64 + k1);
    v4f x1b = *(const v4f*)(x  + row * 64 + k1 + 4);
    v4f n1a = *(const v4f*)(Nh + row * 64 + k1);
    v4f n1b = *(const v4f*)(Nh + row * 64 + k1 + 4);
    v16bf aS, aP;
#pragma unroll
    for (int i = 0; i < 4; ++i) {
      aS[i]      = f2bf(x0a[i] + n0a[i]);  aP[i]      = f2bf(x0a[i] * n0a[i]);
      aS[i + 4]  = f2bf(x0b[i] + n0b[i]);  aP[i + 4]  = f2bf(x0b[i] * n0b[i]);
      aS[i + 8]  = f2bf(x1a[i] + n1a[i]);  aP[i + 8]  = f2bf(x1a[i] * n1a[i]);
      aS[i + 12] = f2bf(x1b[i] + n1b[i]);  aP[i + 12] = f2bf(x1b[i] * n1b[i]);
    }
    v16bf bm1 = *(const v16bf*)(W1B + ((long long)(ct * 2 + kt)) * 512 + lane * 16);
    v16bf bm2 = *(const v16bf*)(W2B + ((long long)(ct * 2 + kt)) * 512 + lane * 16);
    accS = __builtin_amdgcn_wmma_f32_16x16x32_bf16(false, aS, false, bm1,
                                                   (short)0, accS, false, false);
    accP = __builtin_amdgcn_wmma_f32_16x16x32_bf16(false, aP, false, bm2,
                                                   (short)0, accP, false, false);
  }
  int col = ct * 16 + mlo;
  float bb1 = b1[col], bb2 = b2[col];
  float* so = sOut[wave];
#pragma unroll
  for (int i = 0; i < 8; ++i) {
    float v1 = accS[i] + bb1; v1 = v1 > 0.f ? v1 : KG_NEG_SLOPE * v1;
    float v2 = accP[i] + bb2; v2 = v2 > 0.f ? v2 : KG_NEG_SLOPE * v2;
    so[(half * 8 + i) * 16 + mlo] = v1 + v2;
  }
  int rrow = lane >> 1;
  int c8   = (lane & 1) * 8;
  v4f o0 = *(const v4f*)(so + rrow * 16 + c8);
  v4f o1 = *(const v4f*)(so + rrow * 16 + c8 + 4);
  float* gp = out + (long long)(nb + rrow) * DO + ct * 16 + c8;
  *(v4f*)gp       = o0;
  *(v4f*)(gp + 4) = o1;
}

// L2-normalize rows of ego [N x DO] into all_embed[:, colOff:colOff+DO]
__global__ __launch_bounds__(256)
void k_norm(const float* __restrict__ ego, float* __restrict__ AE, int DO, int colOff) {
  int n    = blockIdx.x * 8 + (threadIdx.x >> 5);
  int lane = threadIdx.x & 31;
  if (n >= KG_N_NODES) return;
  float ss = 0.f;
  for (int k = lane; k < DO; k += 32) { float v = ego[(long long)n * DO + k]; ss += v * v; }
#pragma unroll
  for (int m = 16; m; m >>= 1) ss += __shfl_xor(ss, m, 32);
  float sc = 1.0f / fmaxf(sqrtf(ss), 1e-12f);
  for (int k = lane; k < DO; k += 32)
    AE[(long long)n * KG_OUT_DIM + colOff + k] = ego[(long long)n * DO + k] * sc;
}

__global__ void k_copy0(const float* __restrict__ emb, float* __restrict__ AE, long long n) {
  long long i = (long long)blockIdx.x * blockDim.x + threadIdx.x;
  if (i >= n) return;
  long long row = i >> 6;
  int c = (int)(i & 63);
  AE[row * KG_OUT_DIM + c] = emb[i];
}

// ---------------------------------------------------------------------------
// BPR loss: one wave32 per batch element, atomicAdd (scaled 1/B) into d_out.
// ---------------------------------------------------------------------------
__global__ __launch_bounds__(256)
void k_bpr(const float* __restrict__ AE, const int* __restrict__ uid,
           const int* __restrict__ pid, const int* __restrict__ nid,
           float* __restrict__ out, int B) {
  int b    = blockIdx.x * 8 + (threadIdx.x >> 5);
  int lane = threadIdx.x & 31;
  if (b >= B) return;
  const float* ur = AE + (long long)uid[b] * KG_OUT_DIM;
  const float* pr = AE + (long long)pid[b] * KG_OUT_DIM;
  const float* nr = AE + (long long)nid[b] * KG_OUT_DIM;
  float pos = 0.f, neg = 0.f, l2 = 0.f;
  for (int k = lane; k < KG_OUT_DIM; k += 32) {
    float uv = ur[k], pv = pr[k], nv = nr[k];
    pos += uv * pv;
    neg += uv * nv;
    l2  += uv * uv + pv * pv + nv * nv;
  }
#pragma unroll
  for (int m = 16; m; m >>= 1) {
    pos += __shfl_xor(pos, m, 32);
    neg += __shfl_xor(neg, m, 32);
    l2  += __shfl_xor(l2,  m, 32);
  }
  if (lane == 0) {
    float z  = neg - pos;                                  // softplus(z) = -logsigmoid(pos-neg)
    float sp = fmaxf(z, 0.f) + log1pf(__expf(-fabsf(z)));
    atomicAdd(out, (sp + 1e-5f * 0.5f * l2) / (float)B);
  }
}

// ---------------------------------------------------------------------------
// Host orchestration
// ---------------------------------------------------------------------------
extern "C" void kernel_launch(void* const* d_in, const int* in_sizes, int n_in,
                              void* d_out, int out_size, void* d_ws, size_t ws_size,
                              hipStream_t stream) {
  const int*   src  = (const int*)d_in[0];
  const int*   dst  = (const int*)d_in[1];
  const int*   et   = (const int*)d_in[2];
  const int*   uid  = (const int*)d_in[3];
  const int*   pid  = (const int*)d_in[4];
  const int*   nid  = (const int*)d_in[5];
  const float* emb  = (const float*)d_in[6];
  const float* relE = (const float*)d_in[7];
  const float* W_R  = (const float*)d_in[8];
  const float* W1_0 = (const float*)d_in[9];
  const float* b1_0 = (const float*)d_in[10];
  const float* W2_0 = (const float*)d_in[11];
  const float* b2_0 = (const float*)d_in[12];
  const float* W1_1 = (const float*)d_in[13];
  const float* b1_1 = (const float*)d_in[14];
  const float* W2_1 = (const float*)d_in[15];
  const float* b2_1 = (const float*)d_in[16];

  const int E = in_sizes[0];
  const int B = in_sizes[3];
  const long long N = KG_N_NODES;

  // workspace carve (256B aligned regions)
  char* cur = (char*)d_ws;
  auto carve = [&](size_t bytes) -> void* {
    void* p = (void*)cur;
    cur += (bytes + 255) & ~(size_t)255;
    return p;
  };
  float*    P      = (float*)   carve((size_t)KG_N_REL * N * 64 * 4);   // 1.02 GB
  __bf16*   embB   = (__bf16*)  carve((size_t)N * 64 * 2);
  __bf16*   WBrel  = (__bf16*)  carve((size_t)KG_N_REL * 4096 * 2);
  __bf16*   W1B0   = (__bf16*)  carve(4096 * 2);
  __bf16*   W2B0   = (__bf16*)  carve(4096 * 2);
  __bf16*   W1B1   = (__bf16*)  carve(2048 * 2);
  __bf16*   W2B1   = (__bf16*)  carve(2048 * 2);
  float*    att    = (float*)   carve((size_t)E * 4);
  unsigned* segmax = (unsigned*)carve((size_t)N * 4);
  float*    segsum = (float*)   carve((size_t)N * 4);
  float*    Nh     = (float*)   carve((size_t)N * 64 * 4);
  float*    ego1   = (float*)   carve((size_t)N * 64 * 4);
  float*    ego2   = (float*)   carve((size_t)N * 32 * 4);
  float*    AE     = (float*)   carve((size_t)N * KG_OUT_DIM * 4);

  auto blocks = [](long long n, int bs) { return (unsigned)((n + bs - 1) / bs); };

  // init (harness does not re-zero between graph replays)
  k_zero_f32<<<1, 32, 0, stream>>>((float*)d_out, 1);
  k_zero_f32<<<blocks(N, 256), 256, 0, stream>>>((float*)segmax, N);
  k_zero_f32<<<blocks(N, 256), 256, 0, stream>>>(segsum, N);

  // bf16 conversions + B-operand pre-swizzle
  k_f32_to_bf16<<<blocks(N * 64, 256), 256, 0, stream>>>(emb, embB, N * 64);
  k_pack_b<<<blocks((long long)KG_N_REL * 4096, 256), 256, 0, stream>>>(W_R, WBrel, 64, KG_N_REL);
  k_pack_b<<<blocks(4096, 256), 256, 0, stream>>>(W1_0, W1B0, 64, 1);
  k_pack_b<<<blocks(4096, 256), 256, 0, stream>>>(W2_0, W2B0, 64, 1);
  k_pack_b<<<blocks(2048, 256), 256, 0, stream>>>(W1_1, W1B1, 32, 1);
  k_pack_b<<<blocks(2048, 256), 256, 0, stream>>>(W2_1, W2B1, 32, 1);

  // P[r] = emb @ W_R[r]   (WMMA, LDS-staged)
  k_proj<<<dim3(3125, KG_N_REL), 256, 0, stream>>>(embB, WBrel, P);

  // attention + edge softmax
  k_att<<<blocks(E, 8), 256, 0, stream>>>(P, relE, src, dst, et, att, segmax, E);
  k_expsum<<<blocks(E, 256), 256, 0, stream>>>(att, segmax, dst, segsum, E);

  // layer 0
  k_zero_f32<<<blocks(N * 64, 256), 256, 0, stream>>>(Nh, N * 64);
  k_msg<<<blocks(E, 8), 256, 0, stream>>>(emb, att, segsum, src, dst, Nh, E);
  k_bi<64><<<dim3(3125), 256, 0, stream>>>(emb, Nh, W1B0, b1_0, W2B0, b2_0, ego1);
  k_norm<<<blocks(N, 8), 256, 0, stream>>>(ego1, AE, 64, 64);

  // layer 1 (uses un-normalized ego1)
  k_zero_f32<<<blocks(N * 64, 256), 256, 0, stream>>>(Nh, N * 64);
  k_msg<<<blocks(E, 8), 256, 0, stream>>>(ego1, att, segsum, src, dst, Nh, E);
  k_bi<32><<<dim3(1563), 256, 0, stream>>>(ego1, Nh, W1B1, b1_1, W2B1, b2_1, ego2);
  k_norm<<<blocks(N, 8), 256, 0, stream>>>(ego2, AE, 32, 128);

  // all_embed[:, :64] = raw embeddings
  k_copy0<<<blocks(N * 64, 256), 256, 0, stream>>>(emb, AE, N * 64);

  // BPR loss -> d_out[0]
  k_bpr<<<blocks(B, 8), 256, 0, stream>>>(AE, uid, pid, nid, (float*)d_out, B);
}
